// MapperNetwork_84670985273498
// MI455X (gfx1250) — compile-verified, matrix operand order checked by reading, not run on hardware
//
#include <hip/hip_runtime.h>

#define B_ROWS 8192
#define DIN 64
#define DMID 1024
#define DOUT 64
#define NDOM 4
#define TMAX32 264          // 32-row tiles: worst case 256+3=259, rounded to 8-wave multiple
#define PADN (TMAX32 * 32)  // 8448 padded rows

typedef __bf16 bf16_t;
typedef __bf16 v16bf __attribute__((ext_vector_type(16)));
typedef __bf16 v8bf  __attribute__((ext_vector_type(8)));
typedef float  v8f   __attribute__((ext_vector_type(8)));

// ---------------------------------------------------------------------------
// Routing: bucket rows by domain, pad each bucket to a multiple of 32 rows so
// every 32-row wave tile has a single domain (weight matrix).
// ---------------------------------------------------------------------------
__global__ void routing_init(int* __restrict__ counts, int* __restrict__ perm,
                             int* __restrict__ tileDom) {
    int t = blockIdx.x * blockDim.x + threadIdx.x;
    if (t < PADN) perm[t] = -1;
    if (t < TMAX32) tileDom[t] = 0;
    if (t < NDOM) counts[t] = 0;
}

__global__ void routing_count(const int* __restrict__ domains, int* __restrict__ counts) {
    int b = blockIdx.x * blockDim.x + threadIdx.x;
    if (b < B_ROWS) atomicAdd(&counts[domains[b] & 3], 1);
}

__global__ void routing_offsets(const int* __restrict__ counts, int* __restrict__ cursor,
                                int* __restrict__ tileDom) {
    if (threadIdx.x == 0 && blockIdx.x == 0) {
        int s = 0;
        for (int k = 0; k < NDOM; ++k) {
            cursor[k] = s;
            int padded = ((counts[k] + 31) >> 5) << 5;
            for (int t = (s >> 5); t < ((s + padded) >> 5); ++t) tileDom[t] = k;
            s += padded;
        }
    }
}

__global__ void routing_scatter(const int* __restrict__ domains, int* __restrict__ cursor,
                                int* __restrict__ perm) {
    int b = blockIdx.x * blockDim.x + threadIdx.x;
    if (b < B_ROWS) {
        int d = domains[b] & 3;
        int pos = atomicAdd(&cursor[d], 1);
        perm[pos] = b;   // row order within a domain does not affect the math
    }
}

// ---------------------------------------------------------------------------
// Weight pre-pack: fp32 row-major [Kd,N] (optionally batched over NDOM) ->
// bf16 B-operand fragments. Fragment f = kt*(N/16)+nt holds 512 bf16:
// lane l (0..31), elem j (0..15):  k = kt*32 + 16*(j>>3) + 8*(l>>4) + (j&7),
//                                  n = nt*16 + (l&15)
// so each GEMM lane reads its whole v16bf operand as one 32-byte load.
// ---------------------------------------------------------------------------
__global__ void pack_weights(const float* __restrict__ src, bf16_t* __restrict__ dst,
                             int Kd, int N, int batch) {
    size_t e = (size_t)blockIdx.x * blockDim.x + threadIdx.x;
    size_t per = (size_t)Kd * N;
    if (e >= per * (size_t)batch) return;
    int bi = (int)(e / per);
    size_t r = e - (size_t)bi * per;
    int frag = (int)(r >> 9);
    int l    = (int)((r >> 4) & 31);
    int j    = (int)(r & 15);
    int nfr = N >> 4;
    int kt = frag / nfr, nt = frag - kt * nfr;
    int hh = l >> 4, idx = l & 15;
    int k = kt * 32 + ((j >> 3) << 4) + hh * 8 + (j & 7);
    int n = nt * 16 + idx;
    dst[e] = (bf16_t)src[(size_t)bi * per + (size_t)k * N + n];
}

// ---------------------------------------------------------------------------
// WMMA GEMM:  Out[M,N] = act(A[M,Kd] @ W[Kd,N] + bias)
//   - 256 threads = 8 waves; wave w owns a 32(M) x 64(N) tile:
//     2 A fragments, 4 B fragments (reused by both M halves), 8 accumulators
//     of v_wmma_f32_16x16x32_bf16 per K step (K step = 32).
//   - AF32:    A is fp32 (first layer, converted on the fly), else bf16
//   - GATHER:  input rows indirected through perm (branch layer 1)
//   - SCATTER: fp32 output scattered through perm to d_out (final layer)
// ---------------------------------------------------------------------------
template <bool AF32, bool GATHER, bool RELU, bool SCATTER>
__global__ __launch_bounds__(256)
void gemm_bf16_wmma(const void* __restrict__ Av,
                    const bf16_t* __restrict__ Wp,
                    const float* __restrict__ bias,
                    void* __restrict__ Outv,
                    const int* __restrict__ perm,
                    const int* __restrict__ tileDom,
                    int Kd, int N) {
    const int lane = threadIdx.x & 31;
    const int wv   = threadIdx.x >> 5;
    const int hh   = lane >> 4;     // lane half selects K sub-chunks (ISA A/B layout)
    const int idx  = lane & 15;     // row (A) / col (B,C,D) within 16-tile
    const int tile32  = blockIdx.x * 8 + wv;
    const int rowBase = tile32 * 32;
    const int colBase = blockIdx.y * 64;
    const int nfr    = N >> 4;
    const int kIters = Kd >> 5;

    int dom = 0;
    if (tileDom) dom = tileDom[tile32];
    const bf16_t* __restrict__ W  = Wp + (size_t)dom * ((size_t)kIters * nfr * 512);
    const float*  __restrict__ bv = bias + (size_t)dom * N;

    int rowA0 = rowBase + idx;
    int rowA1 = rowBase + 16 + idx;
    if (GATHER) {
        int p0 = perm[rowA0]; rowA0 = (p0 < 0) ? 0 : p0;
        int p1 = perm[rowA1]; rowA1 = (p1 < 0) ? 0 : p1;
    }

    const bf16_t* Wl = W + ((size_t)(colBase >> 4)) * 512 + lane * 16;

    v8f acc0 = {}, acc1 = {}, acc2 = {}, acc3 = {};
    v8f acc4 = {}, acc5 = {}, acc6 = {}, acc7 = {};

    for (int kk = 0; kk < kIters; ++kk) {
        v16bf a0, a1;
        if (AF32) {
            const float* Ap0 = (const float*)Av + (size_t)rowA0 * Kd + kk * 32 + hh * 8;
            const float* Ap1 = (const float*)Av + (size_t)rowA1 * Kd + kk * 32 + hh * 8;
            v8f l0 = *(const v8f*)Ap0;  v8f h0 = *(const v8f*)(Ap0 + 16);
            v8f l1 = *(const v8f*)Ap1;  v8f h1 = *(const v8f*)(Ap1 + 16);
#pragma unroll
            for (int j = 0; j < 8; ++j) {
                a0[j] = (bf16_t)l0[j]; a0[j + 8] = (bf16_t)h0[j];
                a1[j] = (bf16_t)l1[j]; a1[j + 8] = (bf16_t)h1[j];
            }
        } else {
            const bf16_t* Ap0 = (const bf16_t*)Av + (size_t)rowA0 * Kd + kk * 32 + hh * 8;
            const bf16_t* Ap1 = (const bf16_t*)Av + (size_t)rowA1 * Kd + kk * 32 + hh * 8;
            v8bf l0 = *(const v8bf*)Ap0;          // k = kb + 8*hh + [0..7]
            v8bf h0 = *(const v8bf*)(Ap0 + 16);   // k = kb + 16 + 8*hh + [0..7]
            v8bf l1 = *(const v8bf*)Ap1;
            v8bf h1 = *(const v8bf*)(Ap1 + 16);
            a0 = __builtin_shufflevector(l0, h0, 0, 1, 2, 3, 4, 5, 6, 7,
                                                 8, 9, 10, 11, 12, 13, 14, 15);
            a1 = __builtin_shufflevector(l1, h1, 0, 1, 2, 3, 4, 5, 6, 7,
                                                 8, 9, 10, 11, 12, 13, 14, 15);
        }
        const bf16_t* wrow = Wl + (size_t)kk * nfr * 512;
        v16bf b0 = *(const v16bf*)(wrow);
        v16bf b1 = *(const v16bf*)(wrow + 512);
        v16bf b2 = *(const v16bf*)(wrow + 1024);
        v16bf b3 = *(const v16bf*)(wrow + 1536);
        acc0 = __builtin_amdgcn_wmma_f32_16x16x32_bf16(false, a0, false, b0, (short)0, acc0, false, false);
        acc1 = __builtin_amdgcn_wmma_f32_16x16x32_bf16(false, a0, false, b1, (short)0, acc1, false, false);
        acc2 = __builtin_amdgcn_wmma_f32_16x16x32_bf16(false, a0, false, b2, (short)0, acc2, false, false);
        acc3 = __builtin_amdgcn_wmma_f32_16x16x32_bf16(false, a0, false, b3, (short)0, acc3, false, false);
        acc4 = __builtin_amdgcn_wmma_f32_16x16x32_bf16(false, a1, false, b0, (short)0, acc4, false, false);
        acc5 = __builtin_amdgcn_wmma_f32_16x16x32_bf16(false, a1, false, b1, (short)0, acc5, false, false);
        acc6 = __builtin_amdgcn_wmma_f32_16x16x32_bf16(false, a1, false, b2, (short)0, acc6, false, false);
        acc7 = __builtin_amdgcn_wmma_f32_16x16x32_bf16(false, a1, false, b3, (short)0, acc7, false, false);
    }

    v8f accs[8] = {acc0, acc1, acc2, acc3, acc4, acc5, acc6, acc7};
#pragma unroll
    for (int c = 0; c < 4; ++c) {
        int col = colBase + c * 16 + idx;
        float bb = bv[col];
#pragma unroll
        for (int g = 0; g < 2; ++g) {       // two 16-row M halves of the wave tile
#pragma unroll
            for (int r = 0; r < 8; ++r) {
                float v = accs[g * 4 + c][r] + bb;
                if (RELU) v = fmaxf(v, 0.0f);
                int row = rowBase + g * 16 + r + hh * 8;  // C/D layout: VGPR r -> M=r (+8 hi half)
                if (SCATTER) {
                    int orow = perm[row];
                    if (orow >= 0) ((float*)Outv)[(size_t)orow * N + col] = v;
                } else {
                    ((bf16_t*)Outv)[(size_t)row * N + col] = (bf16_t)v;
                }
            }
        }
    }
}

// ---------------------------------------------------------------------------
extern "C" void kernel_launch(void* const* d_in, const int* in_sizes, int n_in,
                              void* d_out, int out_size, void* d_ws, size_t ws_size,
                              hipStream_t stream) {
    (void)in_sizes; (void)n_in; (void)out_size; (void)ws_size;
    const float* X       = (const float*)d_in[0];
    const int*   domains = (const int*)d_in[1];
    const float* W1 = (const float*)d_in[2];  const float* b1 = (const float*)d_in[3];
    const float* W2 = (const float*)d_in[4];  const float* b2 = (const float*)d_in[5];
    const float* W3 = (const float*)d_in[6];  const float* b3 = (const float*)d_in[7];
    const float* W4 = (const float*)d_in[8];  const float* b4 = (const float*)d_in[9];
    const float* BW1 = (const float*)d_in[10]; const float* Bb1 = (const float*)d_in[11];
    const float* BW2 = (const float*)d_in[12]; const float* Bb2 = (const float*)d_in[13];
    const float* BW3 = (const float*)d_in[14]; const float* Bb3 = (const float*)d_in[15];
    const float* BW4 = (const float*)d_in[16]; const float* Bb4 = (const float*)d_in[17];
    float* out = (float*)d_out;
    char*  ws  = (char*)d_ws;

    // ---- workspace layout ----
    int* counts  = (int*)(ws + 0);
    int* cursor  = (int*)(ws + 64);
    int* perm    = (int*)(ws + 4096);
    int* tileDom = (int*)(ws + 4096 + (size_t)PADN * sizeof(int));
    size_t o = 65536;
    const size_t actBytes = (size_t)PADN * DMID * sizeof(bf16_t);
    bf16_t* hA = (bf16_t*)(ws + o); o += actBytes;
    bf16_t* hB = (bf16_t*)(ws + o); o += actBytes;
    bf16_t* W1p  = (bf16_t*)(ws + o); o += (size_t)DIN * DMID * 2;
    bf16_t* W2p  = (bf16_t*)(ws + o); o += (size_t)DMID * DMID * 2;
    bf16_t* W3p  = (bf16_t*)(ws + o); o += (size_t)DMID * DMID * 2;
    bf16_t* W4p  = (bf16_t*)(ws + o); o += (size_t)DMID * DMID * 2;
    bf16_t* BW1p = (bf16_t*)(ws + o); o += (size_t)NDOM * DMID * DMID * 2;
    bf16_t* BW2p = (bf16_t*)(ws + o); o += (size_t)NDOM * DMID * DMID * 2;
    bf16_t* BW3p = (bf16_t*)(ws + o); o += (size_t)NDOM * DMID * DMID * 2;
    bf16_t* BW4p = (bf16_t*)(ws + o); o += (size_t)NDOM * DMID * DOUT * 2;

    // ---- routing metadata ----
    routing_init<<<(PADN + 255) / 256, 256, 0, stream>>>(counts, perm, tileDom);
    routing_count<<<B_ROWS / 256, 256, 0, stream>>>(domains, counts);
    routing_offsets<<<1, 1, 0, stream>>>(counts, cursor, tileDom);
    routing_scatter<<<B_ROWS / 256, 256, 0, stream>>>(domains, cursor, perm);

    // ---- weight packing (fp32 -> bf16 WMMA B-fragments) ----
    auto pack = [&](const float* s, bf16_t* d, int Kd, int N, int batch) {
        size_t tot = (size_t)Kd * N * batch;
        pack_weights<<<(int)((tot + 255) / 256), 256, 0, stream>>>(s, d, Kd, N, batch);
    };
    pack(W1, W1p, DIN, DMID, 1);
    pack(W2, W2p, DMID, DMID, 1);
    pack(W3, W3p, DMID, DMID, 1);
    pack(W4, W4p, DMID, DMID, 1);
    pack(BW1, BW1p, DMID, DMID, NDOM);
    pack(BW2, BW2p, DMID, DMID, NDOM);
    pack(BW3, BW3p, DMID, DMID, NDOM);
    pack(BW4, BW4p, DMID, DOUT, NDOM);

    // ---- shared MLP: 4x (GEMM + bias + ReLU), WG tile = 256(M) x 64(N) ----
    dim3 blk(256, 1, 1);
    dim3 gS(B_ROWS / 256, DMID / 64, 1);   // 32 x 16 workgroups
    gemm_bf16_wmma<true,  false, true, false><<<gS, blk, 0, stream>>>(X,  W1p, b1, hA, nullptr, nullptr, DIN,  DMID);
    gemm_bf16_wmma<false, false, true, false><<<gS, blk, 0, stream>>>(hA, W2p, b2, hB, nullptr, nullptr, DMID, DMID);
    gemm_bf16_wmma<false, false, true, false><<<gS, blk, 0, stream>>>(hB, W3p, b3, hA, nullptr, nullptr, DMID, DMID);
    gemm_bf16_wmma<false, false, true, false><<<gS, blk, 0, stream>>>(hA, W4p, b4, hB, nullptr, nullptr, DMID, DMID);

    // ---- routed branch MLP (each row computed with its own domain only) ----
    dim3 gB(TMAX32 / 8, DMID / 64, 1);     // 33 x 16 workgroups, domain-uniform 32-row tiles
    gemm_bf16_wmma<false, true,  true, false><<<gB, blk, 0, stream>>>(hB, BW1p, Bb1, hA, perm,    tileDom, DMID, DMID);
    gemm_bf16_wmma<false, false, true, false><<<gB, blk, 0, stream>>>(hA, BW2p, Bb2, hB, nullptr, tileDom, DMID, DMID);
    gemm_bf16_wmma<false, false, true, false><<<gB, blk, 0, stream>>>(hB, BW3p, Bb3, hA, nullptr, tileDom, DMID, DMID);
    dim3 gO(TMAX32 / 8, DOUT / 64, 1);     // 33 x 1
    gemm_bf16_wmma<false, false, false, true><<<gO, blk, 0, stream>>>(hA, BW4p, Bb4, out, perm,   tileDom, DMID, DOUT);
}
